// EuclideanGATLayer_76562087019045
// MI455X (gfx1250) — matrix operand order repaired and checked by reading, not compiled, not run on hardware
//
#include <hip/hip_runtime.h>
#include <hip/hip_bf16.h>

typedef __attribute__((ext_vector_type(16))) _Float16 v16h;
typedef __attribute__((ext_vector_type(8)))  float    v8f;
typedef __attribute__((ext_vector_type(4)))  int      v4i;
typedef __attribute__((ext_vector_type(4)))  float    v4f;

#define GAT_N      8192
#define GAT_INDIM  128
#define GAT_F      64
#define TILE_M     16
#define NWAVES     4
#define KCHUNK     32
#define NEG_INF_F  (-9.0e15f)

// ---------------------------------------------------------------------------
// Kernel 1: Wh = h @ W  (store f32 for edge-logit precision, f16 for WMMA B)
// ---------------------------------------------------------------------------
__global__ __launch_bounds__(256) void gat_prep_wh(
    const float* __restrict__ h, const float* __restrict__ W,
    float* __restrict__ Wh, _Float16* __restrict__ Whh)
{
    int idx = blockIdx.x * 256 + threadIdx.x;      // idx = row*64 + n
    int row = idx >> 6;
    int n   = idx & 63;
    const float* hr = h + (size_t)row * GAT_INDIM;
    float acc = 0.f;
#pragma unroll 8
    for (int k = 0; k < GAT_INDIM; ++k)
        acc = fmaf(hr[k], W[k * GAT_F + n], acc);
    Wh[idx]  = acc;
    Whh[idx] = (_Float16)acc;
}

// ---------------------------------------------------------------------------
// Kernel 2: e_src = Wh @ a[:64], e_dst = Wh @ a[64:]
// ---------------------------------------------------------------------------
__global__ __launch_bounds__(256) void gat_prep_e(
    const float* __restrict__ Wh, const float* __restrict__ a,
    float* __restrict__ esrc, float* __restrict__ edst)
{
    int row = blockIdx.x * 256 + threadIdx.x;
    const float* wr = Wh + (size_t)row * GAT_F;
    float s1 = 0.f, s2 = 0.f;
#pragma unroll 8
    for (int n = 0; n < GAT_F; ++n) {
        float v = wr[n];
        s1 = fmaf(v, a[n], s1);
        s2 = fmaf(v, a[GAT_F + n], s2);
    }
    esrc[row] = s1;
    edst[row] = s2;
}

// ---------------------------------------------------------------------------
// Masked edge logits for the 16 elements a lane owns in the WMMA A layout:
//   elements 0..7  -> j0..j0+7,  elements 8..15 -> j1..j1+7
// ---------------------------------------------------------------------------
__device__ __forceinline__ void masked_e16(
    const int* __restrict__ adjp, const float* __restrict__ edst,
    long long arow, int j0, int j1, float es, float ev[16])
{
    v4i a0 = *(const v4i*)(adjp + arow + j0);
    v4i a1 = *(const v4i*)(adjp + arow + j0 + 4);
    v4i a2 = *(const v4i*)(adjp + arow + j1);
    v4i a3 = *(const v4i*)(adjp + arow + j1 + 4);
    v4f d0 = *(const v4f*)(edst + j0);
    v4f d1 = *(const v4f*)(edst + j0 + 4);
    v4f d2 = *(const v4f*)(edst + j1);
    v4f d3 = *(const v4f*)(edst + j1 + 4);
#pragma unroll
    for (int c = 0; c < 4; ++c) {
        float e;
        e = es + d0[c]; e = e > 0.f ? e : 0.01f * e; ev[c]      = a0[c] > 0 ? e : NEG_INF_F;
        e = es + d1[c]; e = e > 0.f ? e : 0.01f * e; ev[4 + c]  = a1[c] > 0 ? e : NEG_INF_F;
        e = es + d2[c]; e = e > 0.f ? e : 0.01f * e; ev[8 + c]  = a2[c] > 0 ? e : NEG_INF_F;
        e = es + d3[c]; e = e > 0.f ? e : 0.01f * e; ev[12 + c] = a3[c] > 0 ? e : NEG_INF_F;
    }
}

// ---------------------------------------------------------------------------
// Kernel 3: single-pass (flash-style) fused masked softmax + attention@Wh + ELU.
// One workgroup (4 waves) per 16-row tile; waves split the j-range; adj is
// read from HBM exactly once. Running per-row max with accumulator rescale
// (wave-uniform skip via ballot so EXEC stays all-ones for WMMA).
// ---------------------------------------------------------------------------
__global__ __launch_bounds__(NWAVES * 32) void gat_main(
    const int* __restrict__ adj, const _Float16* __restrict__ Whh,
    const float* __restrict__ esrc, const float* __restrict__ edst,
    float* __restrict__ out)
{
    __shared__ float s_m[NWAVES][TILE_M];
    __shared__ float s_s[NWAVES][TILE_M];
    __shared__ float s_wscale[NWAVES][TILE_M];
    __shared__ float s_invZ[TILE_M];
    __shared__ float s_d[NWAVES][TILE_M][GAT_F];

    const int tid  = threadIdx.x;
    const int wave = tid >> 5;
    const int lane = tid & 31;
    const int row  = lane & 15;    // m within tile
    const int g    = lane >> 4;    // K-group for A layout / D row-half
    const int m0   = blockIdx.x * TILE_M;
    const int grow = m0 + row;

    const float es = esrc[grow];
    const long long arow = (long long)grow * GAT_N;

    float mrun = NEG_INF_F;
    float srun = 0.f;
    v8f c0 = {}, c1 = {}, c2 = {}, c3 = {};

    for (int jb = wave * KCHUNK; jb < GAT_N; jb += NWAVES * KCHUNK) {
        const int j0 = jb + 8 * g;
        const int j1 = jb + 16 + 8 * g;

        // B tiles first: lane holds Wh column (lane&15) over rows jb+k in A's k order; exp chain below hides latency
        const _Float16* bp = Whh + (long long)jb * GAT_F + (lane & 15);
        v16h B0, B1, B2, B3;
#pragma unroll
        for (int s = 0; s < 16; ++s) {
            const int k = (s < 8) ? (8 * g + s) : (16 + 8 * g + (s - 8));
            B0[s] = bp[k * GAT_F];
            B1[s] = bp[k * GAT_F + 16];
            B2[s] = bp[k * GAT_F + 32];
            B3[s] = bp[k * GAT_F + 48];
        }

        if (jb + NWAVES * KCHUNK < GAT_N)
            __builtin_prefetch(adj + arow + j0 + NWAVES * KCHUNK, 0, 3);

        float ev[16];
        masked_e16(adj, edst, arow, j0, j1, es, ev);

        // chunk row-max (merge the two lanes sharing a row)
        float cmax = ev[0];
#pragma unroll
        for (int c = 1; c < 16; ++c) cmax = fmaxf(cmax, ev[c]);
        cmax = fmaxf(cmax, __shfl_xor(cmax, 16, 32));
        const float mnew = fmaxf(mrun, cmax);

        // rescale running state when any row's max advanced (wave-uniform branch)
        if (__ballot(mnew > mrun)) {
            const float scale = __expf(mrun - mnew);   // <=1; ==1 for unchanged rows
            srun *= scale;
#pragma unroll
            for (int v = 0; v < 8; ++v) {
                // D row of element v is (v + 8*g); fetch that row's scale
                const float sc = __shfl(scale, v + 8 * g, 32);
                c0[v] *= sc; c1[v] *= sc; c2[v] *= sc; c3[v] *= sc;
            }
        }
        mrun = mnew;

        // p = exp(e - m): feeds both the softmax denominator and the WMMA A tile
        v16h A;
        float psum = 0.f;
#pragma unroll
        for (int c = 0; c < 16; ++c) {
            const float p = __expf(ev[c] - mrun);      // masked -> exp(-huge) = 0
            psum += p;
            A[c] = (_Float16)p;
        }
        srun += psum;

        c0 = __builtin_amdgcn_wmma_f32_16x16x32_f16(false, A, false, B0, (short)0, c0, false, false);
        c1 = __builtin_amdgcn_wmma_f32_16x16x32_f16(false, A, false, B1, (short)0, c1, false, false);
        c2 = __builtin_amdgcn_wmma_f32_16x16x32_f16(false, A, false, B2, (short)0, c2, false, false);
        c3 = __builtin_amdgcn_wmma_f32_16x16x32_f16(false, A, false, B3, (short)0, c3, false, false);
    }

    // merge the two lanes sharing a row (same mrun on both, synced every chunk)
    srun += __shfl_xor(srun, 16, 32);
    if (lane < 16) { s_m[wave][row] = mrun; s_s[wave][row] = srun; }
    __syncthreads();

    // cross-wave softmax merge: global max, per-wave rescale factors, 1/Z
    if (tid < TILE_M) {
        float m = s_m[0][tid];
#pragma unroll
        for (int w = 1; w < NWAVES; ++w) m = fmaxf(m, s_m[w][tid]);
        float z = 0.f;
#pragma unroll
        for (int w = 0; w < NWAVES; ++w) {
            const float sc = __expf(s_m[w][tid] - m);
            s_wscale[w][tid] = sc;
            z += s_s[w][tid] * sc;
        }
        s_invZ[tid] = 1.0f / z;
    }

    // spill per-wave partial D to LDS:  D row = v + 8*g, col = lane&15
    const int col = lane & 15;
#pragma unroll
    for (int v = 0; v < 8; ++v) {
        const int m = v + 8 * g;
        s_d[wave][m][0  + col] = c0[v];
        s_d[wave][m][16 + col] = c1[v];
        s_d[wave][m][32 + col] = c2[v];
        s_d[wave][m][48 + col] = c3[v];
    }
    __syncthreads();

    // reduce waves (with per-wave softmax rescale), normalize, ELU, store
    for (int idx = tid; idx < TILE_M * GAT_F; idx += NWAVES * 32) {
        const int m = idx >> 6;
        const int n = idx & 63;
        float acc = 0.f;
#pragma unroll
        for (int w = 0; w < NWAVES; ++w)
            acc = fmaf(s_d[w][m][n], s_wscale[w][m], acc);
        acc *= s_invZ[m];
        acc = acc > 0.f ? acc : (__expf(acc) - 1.0f);
        out[(long long)(m0 + m) * GAT_F + n] = acc;
    }
}

// ---------------------------------------------------------------------------
extern "C" void kernel_launch(void* const* d_in, const int* in_sizes, int n_in,
                              void* d_out, int out_size, void* d_ws, size_t ws_size,
                              hipStream_t stream)
{
    const float* h   = (const float*)d_in[0];
    const int*   adj = (const int*)d_in[1];
    const float* W   = (const float*)d_in[2];
    const float* a   = (const float*)d_in[3];
    float* out = (float*)d_out;

    char* ws = (char*)d_ws;
    float*    Wh   = (float*)ws;                                      // 8192*64 f32 (2 MB)
    _Float16* Whh  = (_Float16*)(ws + (size_t)GAT_N * GAT_F * 4);     // 8192*64 f16 (1 MB)
    float*    esrc = (float*)(ws + (size_t)GAT_N * GAT_F * 6);        // 32 KB
    float*    edst = esrc + GAT_N;                                    // 32 KB

    gat_prep_wh<<<(GAT_N * GAT_F) / 256, 256, 0, stream>>>(h, W, Wh, Whh);
    gat_prep_e <<<GAT_N / 256, 256, 0, stream>>>(Wh, a, esrc, edst);
    gat_main   <<<GAT_N / TILE_M, NWAVES * 32, 0, stream>>>(adj, Whh, esrc, edst, out);
}
